// DeepSeekV2StyleMoE_71640054497667
// MI455X (gfx1250) — matrix operand order repaired and checked by reading, not compiled, not run on hardware
//
#include <hip/hip_runtime.h>
#include <hip/hip_bf16.h>
#include <stdint.h>

// ---------------- problem constants ----------------
#define T_TOK   2048      // B*S tokens
#define HDIM    2048
#define FDIM    1408
#define FSDIM   2816
#define NEXP    8
#define TOPK    2

// ---------------- tiling ----------------
#define BM      128       // block M tile = 8 waves x 16 rows
#define BN      64        // block N tile = 4 wmma tiles / wave
#define KC      64        // K chunk staged in LDS
#define BLK     256       // 8 wave32s
#define NWAVES  8

typedef __bf16 bf16_t;
typedef __bf16 v16bf __attribute__((ext_vector_type(16)));
typedef float  v8f   __attribute__((ext_vector_type(8)));
typedef unsigned int v4u __attribute__((ext_vector_type(4)));
typedef int v4i __attribute__((ext_vector_type(4)));
typedef int v8i __attribute__((ext_vector_type(8)));

union Frag16 { v4i q[2]; v16bf v; };   // 32 bytes = 16 bf16 = one A/B fragment per lane

// 16-bit A 16x32 fragment: lane m (0-15) holds M=m, K={kb..kb+7, kb+16..kb+23}
// with kb = k + (lane>=16 ? 8 : 0)  (ISA 7.12.2 table)
__device__ __forceinline__ v16bf load_a_frag(const bf16_t* rowBase, int kb) {
  Frag16 f;
  f.q[0] = *(const v4i*)(rowBase + kb);
  f.q[1] = *(const v4i*)(rowBase + kb + 16);
  return f.v;
}

__device__ __forceinline__ v8f wmma_bf16(v16bf a, v16bf b, v8f c) {
  // D = A(16x32) * B(32x16) + C ; f32 accumulate
  return __builtin_amdgcn_wmma_f32_16x16x32_bf16(false, a, false, b, (short)0, c,
                                                 false, false);
}

// ---------------- scheduler shaping ----------------
// Group per k-step: VMEM reads (A) -> DS reads (B frags) -> WMMAs, so one wait
// covers a clause of LDS loads instead of a ds_load/wait/wmma ping-pong.
#if defined(__AMDGCN__) && __has_builtin(__builtin_amdgcn_sched_group_barrier)
#define SCHED_GROUP(mask, size) __builtin_amdgcn_sched_group_barrier((mask), (size), 0)
#else
#define SCHED_GROUP(mask, size)
#endif
// masks: 0x020 = VMEM read, 0x100 = DS read, 0x008 = MFMA/WMMA

// ---------------- Tensor Data Mover (gfx1250) ----------------
#if defined(__AMDGCN__) && __has_builtin(__builtin_amdgcn_tensor_load_to_lds)
#define USE_TDM 1
#endif

#ifdef USE_TDM
// 2D tile load: tileK contiguous 2-byte elements (dim0) x tileRows lines.
// D# bitfields per CDNA5 ISA (group0 128b, group1 256b).
__device__ __forceinline__ void tdm_load_tile_2d(const bf16_t* gptr, unsigned lds_off,
                                                 unsigned tileK, unsigned tileRows,
                                                 unsigned dim0, unsigned dim1,
                                                 unsigned long long stride0) {
  unsigned long long ga = (unsigned long long)(uintptr_t)gptr;
  v4u g0;
  g0.x = 1u;                                                   // count=1, user D#
  g0.y = lds_off;                                              // lds_addr
  g0.z = (unsigned)ga;                                         // global_addr[31:0]
  g0.w = ((unsigned)(ga >> 32) & 0x01FFFFFFu) | 0x80000000u;   // addr[56:32] | type=2
  v8i g1;
  g1[0] = 0x10000;                                             // data_size=1 -> 2 bytes
  g1[1] = (int)((dim0 & 0xFFFFu) << 16);                       // tensor_dim0[15:0]
  g1[2] = (int)(((dim0 >> 16) & 0xFFFFu) | ((dim1 & 0xFFFFu) << 16));
  g1[3] = (int)(((dim1 >> 16) & 0xFFFFu) | (tileK << 16));     // tile_dim0
  g1[4] = (int)tileRows;                                       // tile_dim1 (tile_dim2=0)
  g1[5] = (int)(unsigned)(stride0 & 0xFFFFFFFFull);            // tensor_dim0_stride lo
  g1[6] = (int)(unsigned)((stride0 >> 32) & 0xFFFFull);        // stride hi | dim1_stride=0
  g1[7] = 0;
  v4i gz = {0, 0, 0, 0};
#if defined(__clang_major__) && __clang_major__ >= 23
  v8i z8 = {0, 0, 0, 0, 0, 0, 0, 0};
  __builtin_amdgcn_tensor_load_to_lds(g0, g1, gz, gz, z8, 0);
#else
  __builtin_amdgcn_tensor_load_to_lds(g0, g1, gz, gz, 0);
#endif
}
#if __has_builtin(__builtin_amdgcn_s_wait_tensorcnt)
#define TDM_WAIT(n) __builtin_amdgcn_s_wait_tensorcnt(n)
#else
#define TDM_WAIT(n) asm volatile("s_wait_tensorcnt %0" ::"i"(n) : "memory")
#endif
#endif  // USE_TDM

// =================== f32 -> bf16 conversion ===================
__global__ __launch_bounds__(BLK) void cvt_f32_bf16_kernel(const float* __restrict__ in,
                                                           bf16_t* __restrict__ out,
                                                           long long n) {
  long long i = (long long)blockIdx.x * blockDim.x + threadIdx.x;
  long long stride = (long long)gridDim.x * blockDim.x;
  for (long long j = i * 4; j < n; j += stride * 4) {
    float4 v = *(const float4*)(in + j);
    out[j + 0] = (bf16_t)v.x;
    out[j + 1] = (bf16_t)v.y;
    out[j + 2] = (bf16_t)v.z;
    out[j + 3] = (bf16_t)v.w;
  }
}

// =================== routing: softmax + group-limited top-2 ===================
__global__ __launch_bounds__(BLK) void moe_routing_kernel(const float* __restrict__ h,
                                                          const float* __restrict__ gw,
                                                          int* __restrict__ topk_i,
                                                          float* __restrict__ topk_w) {
  const int wave = threadIdx.x >> 5;
  const int lane = threadIdx.x & 31;
  const int t = blockIdx.x * NWAVES + wave;

  float acc[NEXP];
#pragma unroll
  for (int e = 0; e < NEXP; ++e) acc[e] = 0.f;
  const float* hrow = h + (size_t)t * HDIM;
  for (int k = lane; k < HDIM; k += 32) {
    float x = hrow[k];
#pragma unroll
    for (int e = 0; e < NEXP; ++e) acc[e] += x * gw[e * HDIM + k];
  }
#pragma unroll
  for (int e = 0; e < NEXP; ++e) {
#pragma unroll
    for (int off = 16; off > 0; off >>= 1) acc[e] += __shfl_down(acc[e], off, 32);
  }
  if (lane == 0) {
    float m = acc[0];
#pragma unroll
    for (int e = 1; e < NEXP; ++e) m = fmaxf(m, acc[e]);
    float s[NEXP], sum = 0.f;
#pragma unroll
    for (int e = 0; e < NEXP; ++e) { s[e] = __expf(acc[e] - m); sum += s[e]; }
    float inv = 1.f / sum;
#pragma unroll
    for (int e = 0; e < NEXP; ++e) s[e] *= inv;
    float gs[4];
#pragma unroll
    for (int g = 0; g < 4; ++g) gs[g] = fmaxf(s[2 * g], s[2 * g + 1]);
    int g0 = 0;
#pragma unroll
    for (int g = 1; g < 4; ++g) if (gs[g] > gs[g0]) g0 = g;
    int g1 = (g0 == 0) ? 1 : 0;
#pragma unroll
    for (int g = 0; g < 4; ++g) if (g != g0 && gs[g] > gs[g1]) g1 = g;
    float msk[NEXP];
#pragma unroll
    for (int e = 0; e < NEXP; ++e) {
      int g = e >> 1;
      msk[e] = (g == g0 || g == g1) ? s[e] : 0.f;
    }
    int i0 = 0;
#pragma unroll
    for (int e = 1; e < NEXP; ++e) if (msk[e] > msk[i0]) i0 = e;
    int i1 = (i0 == 0) ? 1 : 0;
#pragma unroll
    for (int e = 0; e < NEXP; ++e) if (e != i0 && msk[e] > msk[i1]) i1 = e;
    topk_i[t * 2 + 0] = i0; topk_w[t * 2 + 0] = msk[i0];   // ROUTED_SCALE = 1
    topk_i[t * 2 + 1] = i1; topk_w[t * 2 + 1] = msk[i1];
  }
}

// =================== dispatch: bucket tokens per expert ===================
__global__ __launch_bounds__(BLK) void moe_dispatch_kernel(const int* __restrict__ topk_i,
                                                           const float* __restrict__ topk_w,
                                                           int* __restrict__ cnt,
                                                           int* __restrict__ basep,
                                                           int* __restrict__ tok_list,
                                                           int* __restrict__ slot_list,
                                                           float* __restrict__ wt_list) {
  __shared__ int sc[NEXP], sb[NEXP], scur[NEXP];
  const int tid = threadIdx.x;
  if (tid < NEXP) { sc[tid] = 0; scur[tid] = 0; }
  __syncthreads();
  for (int i = tid; i < T_TOK * TOPK; i += BLK) atomicAdd(&sc[topk_i[i]], 1);
  __syncthreads();
  if (tid == 0) {
    int o = 0;
    for (int e = 0; e < NEXP; ++e) { sb[e] = o; o += sc[e]; }
  }
  __syncthreads();
  for (int i = tid; i < T_TOK * TOPK; i += BLK) {
    int e = topk_i[i];
    int p = sb[e] + atomicAdd(&scur[e], 1);
    tok_list[p] = i >> 1;
    slot_list[p] = i & 1;
    wt_list[p] = topk_w[i];
  }
  if (tid < NEXP) { cnt[tid] = sc[tid]; basep[tid] = sb[tid]; }
}

// =================== gate+up GEMM: double-buffered TDM weights, fused SiLU =====
// act[row, f] = silu(X@Wg^T) * (X@Wu^T), gathered rows when tok_list != null
__global__ __launch_bounds__(BLK) void moe_gateup_kernel(
    const bf16_t* __restrict__ X,   // [T_TOK, HDIM]
    const bf16_t* __restrict__ Wg,  // [E, Fdim, HDIM]
    const bf16_t* __restrict__ Wu,  // [E, Fdim, HDIM]
    bf16_t* __restrict__ Act,       // [rows, Fdim]
    const int* __restrict__ cnt, const int* __restrict__ basep,
    const int* __restrict__ tok_list, int Fdim, int rowsTotal) {
  const int e = blockIdx.z;
  const int myCnt = cnt ? cnt[e] : rowsTotal;
  const int myBase = basep ? basep[e] : 0;
  const int m0 = blockIdx.x * BM;
  if (m0 >= myCnt) return;
  const int n0 = blockIdx.y * BN;

  const size_t wOff = (size_t)e * Fdim * HDIM;
  const bf16_t* WgE = Wg + wOff;
  const bf16_t* WuE = Wu + wOff;

  __shared__ __align__(16) bf16_t sW[2][2][BN][KC];   // [buf][gate/up] = 32 KB

  const int wave = threadIdx.x >> 5;
  const int lane = threadIdx.x & 31;
  const int half = lane >> 4;
  const int l15 = lane & 15;

  const int aRow = m0 + wave * 16 + l15;
  const int aIdx = (aRow < myCnt) ? aRow : (myCnt - 1);
  const int tok = tok_list ? tok_list[myBase + aIdx] : aIdx;
  const bf16_t* aBase = X + (size_t)tok * HDIM;

  v8f accG[4] = {};
  v8f accU[4] = {};

  // one k-step (K=32) of the 16x64 output strip: batch all 8 B frags, then 8 WMMAs
#define GATEUP_STEP(BUF, KK)                                                     \
  {                                                                              \
    v16bf aF = load_a_frag(aBase, k0 + (KK) + (half << 3));                      \
    const int kb = (KK) + (half << 3);                                           \
    Frag16 bg[4], bu[4];                                                         \
    _Pragma("unroll") for (int j = 0; j < 4; ++j) {                              \
      bg[j].q[0] = *(const v4i*)&sW[BUF][0][j * 16 + l15][kb];                   \
      bg[j].q[1] = *(const v4i*)&sW[BUF][0][j * 16 + l15][kb + 16];              \
      bu[j].q[0] = *(const v4i*)&sW[BUF][1][j * 16 + l15][kb];                   \
      bu[j].q[1] = *(const v4i*)&sW[BUF][1][j * 16 + l15][kb + 16];              \
    }                                                                            \
    _Pragma("unroll") for (int j = 0; j < 4; ++j) {                              \
      accG[j] = wmma_bf16(aF, bg[j].v, accG[j]);                                 \
      accU[j] = wmma_bf16(aF, bu[j].v, accU[j]);                                 \
    }                                                                            \
    SCHED_GROUP(0x020, 2);  /* A: 2 global_load_b128 */                          \
    SCHED_GROUP(0x100, 16); /* B: 16 ds_load_b128 in one clause */               \
    SCHED_GROUP(0x008, 8);  /* 8 back-to-back WMMAs */                           \
  }

#ifdef USE_TDM
  // pipeline prologue: DMA chunk 0 into buffer 0
  if (wave == 0) {
    tdm_load_tile_2d(WgE + (size_t)n0 * HDIM, (unsigned)(uintptr_t)&sW[0][0][0][0],
                     KC, BN, (unsigned)HDIM, (unsigned)Fdim, (unsigned long long)HDIM);
    tdm_load_tile_2d(WuE + (size_t)n0 * HDIM, (unsigned)(uintptr_t)&sW[0][1][0][0],
                     KC, BN, (unsigned)HDIM, (unsigned)Fdim, (unsigned long long)HDIM);
  }
  for (int k0 = 0; k0 < HDIM; k0 += KC) {
    const int cur = (k0 / KC) & 1;
    const int nxt = cur ^ 1;
    if (wave == 0) {
      if (k0 + KC < HDIM) {
        // overlap: DMA chunk i+1 while chunk i computes; done-signals are in-order,
        // so tensorcnt<=2 guarantees chunk i's two tiles have landed.
        tdm_load_tile_2d(WgE + (size_t)n0 * HDIM + (k0 + KC),
                         (unsigned)(uintptr_t)&sW[nxt][0][0][0],
                         KC, BN, (unsigned)HDIM, (unsigned)Fdim,
                         (unsigned long long)HDIM);
        tdm_load_tile_2d(WuE + (size_t)n0 * HDIM + (k0 + KC),
                         (unsigned)(uintptr_t)&sW[nxt][1][0][0],
                         KC, BN, (unsigned)HDIM, (unsigned)Fdim,
                         (unsigned long long)HDIM);
        TDM_WAIT(2);
      } else {
        TDM_WAIT(0);
      }
    }
    __syncthreads();                      // chunk i visible to all waves
    __builtin_prefetch(aBase + k0 + KC, 0, 3);   // next A chunk into near caches
    GATEUP_STEP(cur, 0)
    GATEUP_STEP(cur, 32)
    __syncthreads();                      // all reads of buf done before reuse
  }
#else
  for (int k0 = 0; k0 < HDIM; k0 += KC) {
    __syncthreads();
    for (int i = threadIdx.x; i < BN * (KC / 2); i += BLK) {
      int r = i / (KC / 2), c = i % (KC / 2);
      ((unsigned*)&sW[0][0][r][0])[c] =
          ((const unsigned*)(WgE + (size_t)(n0 + r) * HDIM + k0))[c];
      ((unsigned*)&sW[0][1][r][0])[c] =
          ((const unsigned*)(WuE + (size_t)(n0 + r) * HDIM + k0))[c];
    }
    __syncthreads();
    GATEUP_STEP(0, 0)
    GATEUP_STEP(0, 32)
  }
#endif
#undef GATEUP_STEP

  // C layout: VGPR r -> row = r + (lane<16 ? 0 : 8), col = lane&15
#pragma unroll
  for (int j = 0; j < 4; ++j) {
#pragma unroll
    for (int r = 0; r < 8; ++r) {
      const int row = m0 + wave * 16 + (half << 3) + r;
      if (row < myCnt) {
        float g = accG[j][r];
        float u = accU[j][r];
        float sg = g / (1.0f + __expf(-g));   // silu
        const int col = n0 + j * 16 + l15;
        Act[(size_t)(myBase + row) * Fdim + col] = (bf16_t)(sg * u);
      }
    }
  }
}

// =================== down GEMM: double-buffered TDM weights, scaled scatter ====
__global__ __launch_bounds__(BLK) void moe_down_kernel(
    const bf16_t* __restrict__ Act,  // [rows, Kdim]
    const bf16_t* __restrict__ Wd,   // [E, HDIM, Kdim]
    float* __restrict__ Out,         // routed: [2T, HDIM] ; shared: [T, HDIM]
    const int* __restrict__ cnt, const int* __restrict__ basep,
    const int* __restrict__ tok_list, const int* __restrict__ slot_list,
    const float* __restrict__ wt_list, int Kdim, int rowsTotal) {
  const int e = blockIdx.z;
  const int myCnt = cnt ? cnt[e] : rowsTotal;
  const int myBase = basep ? basep[e] : 0;
  const int m0 = blockIdx.x * BM;
  if (m0 >= myCnt) return;
  const int n0 = blockIdx.y * BN;

  const bf16_t* WdE = Wd + (size_t)e * HDIM * Kdim;

  __shared__ __align__(16) bf16_t sW[2][BN][KC];   // 16 KB double buffer

  const int wave = threadIdx.x >> 5;
  const int lane = threadIdx.x & 31;
  const int half = lane >> 4;
  const int l15 = lane & 15;

  const int aRow = m0 + wave * 16 + l15;
  const int aIdx = (aRow < myCnt) ? aRow : (myCnt - 1);
  const bf16_t* aBase = Act + (size_t)(myBase + aIdx) * Kdim;

  v8f acc[4] = {};

#define DOWN_STEP(BUF, KK)                                                       \
  {                                                                              \
    v16bf aF = load_a_frag(aBase, k0 + (KK) + (half << 3));                      \
    const int kb = (KK) + (half << 3);                                           \
    Frag16 b[4];                                                                 \
    _Pragma("unroll") for (int j = 0; j < 4; ++j) {                              \
      b[j].q[0] = *(const v4i*)&sW[BUF][j * 16 + l15][kb];                       \
      b[j].q[1] = *(const v4i*)&sW[BUF][j * 16 + l15][kb + 16];                  \
    }                                                                            \
    _Pragma("unroll") for (int j = 0; j < 4; ++j) {                              \
      acc[j] = wmma_bf16(aF, b[j].v, acc[j]);                                    \
    }                                                                            \
    SCHED_GROUP(0x020, 2);  /* A: 2 global_load_b128 */                          \
    SCHED_GROUP(0x100, 8);  /* B: 8 ds_load_b128 in one clause */                \
    SCHED_GROUP(0x008, 4);  /* 4 back-to-back WMMAs */                           \
  }

#ifdef USE_TDM
  if (wave == 0) {
    tdm_load_tile_2d(WdE + (size_t)n0 * Kdim, (unsigned)(uintptr_t)&sW[0][0][0],
                     KC, BN, (unsigned)Kdim, (unsigned)HDIM, (unsigned long long)Kdim);
  }
  for (int k0 = 0; k0 < Kdim; k0 += KC) {
    const int cur = (k0 / KC) & 1;
    const int nxt = cur ^ 1;
    if (wave == 0) {
      if (k0 + KC < Kdim) {
        tdm_load_tile_2d(WdE + (size_t)n0 * Kdim + (k0 + KC),
                         (unsigned)(uintptr_t)&sW[nxt][0][0],
                         KC, BN, (unsigned)Kdim, (unsigned)HDIM,
                         (unsigned long long)Kdim);
        TDM_WAIT(1);
      } else {
        TDM_WAIT(0);
      }
    }
    __syncthreads();
    __builtin_prefetch(aBase + k0 + KC, 0, 3);
    DOWN_STEP(cur, 0)
    DOWN_STEP(cur, 32)
    __syncthreads();
  }
#else
  for (int k0 = 0; k0 < Kdim; k0 += KC) {
    __syncthreads();
    for (int i = threadIdx.x; i < BN * (KC / 2); i += BLK) {
      int r = i / (KC / 2), c = i % (KC / 2);
      ((unsigned*)&sW[0][r][0])[c] =
          ((const unsigned*)(WdE + (size_t)(n0 + r) * Kdim + k0))[c];
    }
    __syncthreads();
    DOWN_STEP(0, 0)
    DOWN_STEP(0, 32)
  }
#endif
#undef DOWN_STEP

#pragma unroll
  for (int j = 0; j < 4; ++j) {
#pragma unroll
    for (int r = 0; r < 8; ++r) {
      const int row = m0 + wave * 16 + (half << 3) + r;
      if (row < myCnt) {
        const int col = n0 + j * 16 + l15;
        const float v = acc[j][r];
        if (tok_list) {
          const int gi = myBase + row;
          const int t = tok_list[gi];
          const int s = slot_list[gi];
          Out[((size_t)t * 2 + s) * HDIM + col] = wt_list[gi] * v;
        } else {
          Out[(size_t)row * HDIM + col] = v;
        }
      }
    }
  }
}

// =================== combine: routed slot0 + slot1 + shared ===================
__global__ __launch_bounds__(BLK) void moe_combine_kernel(const float* __restrict__ routed,
                                                          const float* __restrict__ shared_o,
                                                          float* __restrict__ out) {
  const size_t idx4 = ((size_t)blockIdx.x * BLK + threadIdx.x) * 4;
  if (idx4 >= (size_t)T_TOK * HDIM) return;
  const size_t t = idx4 / HDIM;
  const size_t c = idx4 % HDIM;
  float4 r0 = *(const float4*)(routed + ((size_t)t * 2 + 0) * HDIM + c);
  float4 r1 = *(const float4*)(routed + ((size_t)t * 2 + 1) * HDIM + c);
  float4 sh = *(const float4*)(shared_o + idx4);
  float4 o;
  o.x = r0.x + r1.x + sh.x;
  o.y = r0.y + r1.y + sh.y;
  o.z = r0.z + r1.z + sh.z;
  o.w = r0.w + r1.w + sh.w;
  *(float4*)(out + idx4) = o;
}

// =================== host-side launch ===================
static inline void* ws_alloc(char*& p, size_t bytes) {
  void* r = (void*)p;
  p += (bytes + 255) & ~(size_t)255;
  return r;
}

extern "C" void kernel_launch(void* const* d_in, const int* in_sizes, int n_in,
                              void* d_out, int out_size, void* d_ws, size_t ws_size,
                              hipStream_t stream) {
  (void)in_sizes; (void)n_in; (void)out_size; (void)ws_size;
  const float* h     = (const float*)d_in[0];   // [2,1024,2048]
  const float* gw    = (const float*)d_in[1];   // [8,2048]
  const float* gate  = (const float*)d_in[2];   // [8,1408,2048]
  const float* up    = (const float*)d_in[3];
  const float* down  = (const float*)d_in[4];   // [8,2048,1408]
  const float* sgate = (const float*)d_in[5];   // [2816,2048]
  const float* sup   = (const float*)d_in[6];
  const float* sdown = (const float*)d_in[7];   // [2048,2816]
  float* out = (float*)d_out;

  // workspace carve (~255 MB total; bf16 weight copies dominate)
  char* p = (char*)d_ws;
  bf16_t* hB   = (bf16_t*)ws_alloc(p, (size_t)T_TOK * HDIM * 2);
  bf16_t* wgB  = (bf16_t*)ws_alloc(p, (size_t)NEXP * FDIM * HDIM * 2);
  bf16_t* wuB  = (bf16_t*)ws_alloc(p, (size_t)NEXP * FDIM * HDIM * 2);
  bf16_t* wdB  = (bf16_t*)ws_alloc(p, (size_t)NEXP * HDIM * FDIM * 2);
  bf16_t* sgB  = (bf16_t*)ws_alloc(p, (size_t)FSDIM * HDIM * 2);
  bf16_t* suB  = (bf16_t*)ws_alloc(p, (size_t)FSDIM * HDIM * 2);
  bf16_t* sdB  = (bf16_t*)ws_alloc(p, (size_t)HDIM * FSDIM * 2);
  int*    tkI  = (int*)   ws_alloc(p, (size_t)T_TOK * TOPK * 4);
  float*  tkW  = (float*) ws_alloc(p, (size_t)T_TOK * TOPK * 4);
  int*    cnt  = (int*)   ws_alloc(p, NEXP * 4);
  int*    base = (int*)   ws_alloc(p, NEXP * 4);
  int*    tokL = (int*)   ws_alloc(p, (size_t)T_TOK * TOPK * 4);
  int*    sltL = (int*)   ws_alloc(p, (size_t)T_TOK * TOPK * 4);
  float*  wtL  = (float*) ws_alloc(p, (size_t)T_TOK * TOPK * 4);
  bf16_t* actR = (bf16_t*)ws_alloc(p, (size_t)T_TOK * TOPK * FDIM * 2);
  bf16_t* actS = (bf16_t*)ws_alloc(p, (size_t)T_TOK * FSDIM * 2);
  float*  rout = (float*) ws_alloc(p, (size_t)T_TOK * TOPK * HDIM * 4);
  float*  sho  = (float*) ws_alloc(p, (size_t)T_TOK * HDIM * 4);

  // 1) bf16 conversions
  cvt_f32_bf16_kernel<<<2048, BLK, 0, stream>>>(h, hB, (long long)T_TOK * HDIM);
  cvt_f32_bf16_kernel<<<4096, BLK, 0, stream>>>(gate, wgB, (long long)NEXP * FDIM * HDIM);
  cvt_f32_bf16_kernel<<<4096, BLK, 0, stream>>>(up, wuB, (long long)NEXP * FDIM * HDIM);
  cvt_f32_bf16_kernel<<<4096, BLK, 0, stream>>>(down, wdB, (long long)NEXP * HDIM * FDIM);
  cvt_f32_bf16_kernel<<<2048, BLK, 0, stream>>>(sgate, sgB, (long long)FSDIM * HDIM);
  cvt_f32_bf16_kernel<<<2048, BLK, 0, stream>>>(sup, suB, (long long)FSDIM * HDIM);
  cvt_f32_bf16_kernel<<<2048, BLK, 0, stream>>>(sdown, sdB, (long long)HDIM * FSDIM);

  // 2) routing + 3) dispatch
  moe_routing_kernel<<<T_TOK / NWAVES, BLK, 0, stream>>>(h, gw, tkI, tkW);
  moe_dispatch_kernel<<<1, BLK, 0, stream>>>(tkI, tkW, cnt, base, tokL, sltL, wtL);

  // 4) routed experts: gate+up (gathered), then down (scaled scatter)
  moe_gateup_kernel<<<dim3(T_TOK / BM, FDIM / BN, NEXP), BLK, 0, stream>>>(
      hB, wgB, wuB, actR, cnt, base, tokL, FDIM, 0);
  moe_down_kernel<<<dim3(T_TOK / BM, HDIM / BN, NEXP), BLK, 0, stream>>>(
      actR, wdB, rout, cnt, base, tokL, sltL, wtL, FDIM, 0);

  // 5) shared expert: dense (identity gather)
  moe_gateup_kernel<<<dim3(T_TOK / BM, FSDIM / BN, 1), BLK, 0, stream>>>(
      hB, sgB, suB, actS, nullptr, nullptr, nullptr, FSDIM, T_TOK);
  moe_down_kernel<<<dim3(T_TOK / BM, HDIM / BN, 1), BLK, 0, stream>>>(
      actS, sdB, sho, nullptr, nullptr, nullptr, nullptr, nullptr, FSDIM, T_TOK);

  // 6) combine
  moe_combine_kernel<<<(T_TOK * HDIM / 4) / BLK, BLK, 0, stream>>>(rout, sho, out);
}
